// FmFMLayer_65429531787434
// MI455X (gfx1250) — matrix operand order repaired and checked by reading, not compiled, not run on hardware
//
#include <hip/hip_runtime.h>
#include <stdint.h>

#define BATCH   16384
#define FIELDS  24
#define DIM     64
#define NPAIRS  276
#define NSPLIT  4
#define LSTR    72      // LDS row stride in halfwords: 64 + 8 pad (16B-aligned rows, conflict-free)

typedef __attribute__((ext_vector_type(16))) __bf16 v16bf;
typedef __attribute__((ext_vector_type(8)))  float  v8f;

union BfFrag { v16bf v; uint32_t u[8]; uint4 q[2]; };

// pack two f32 -> {bf16(lo),bf16(hi)} by truncation: one v_perm_b32
__device__ __forceinline__ uint32_t pack_bf16_trunc(float lo, float hi) {
    uint32_t a = __builtin_bit_cast(uint32_t, lo);
    uint32_t b = __builtin_bit_cast(uint32_t, hi);
    return __builtin_amdgcn_perm(b, a, 0x07060302u);
}

// async copy 32B (two b128) from global to LDS; completion tracked by ASYNCcnt
__device__ __forceinline__ void async_copy32(uint32_t ldsOfs, const uint16_t* src) {
    uint64_t ga = (uint64_t)(uintptr_t)src;
    asm volatile("global_load_async_to_lds_b128 %0, %1, off\n\t"
                 "global_load_async_to_lds_b128 %0, %1, off offset:16"
                 :: "v"(ldsOfs), "v"(ga) : "memory");
}
__device__ __forceinline__ void wait_async0() {
    asm volatile("s_wait_asynccnt 0x0" ::: "memory");
}

// ---------- prepass: W (276,64,64) f32 row-major  ->  Wt (276,64,64) bf16 [e][d] ----------
__global__ __launch_bounds__(256) void w_transpose_bf16(
    const float* __restrict__ W, uint16_t* __restrict__ Wt)
{
    __shared__ uint16_t t[DIM * 66];
    const int p = blockIdx.x, tid = threadIdx.x;
    const float* wp = W + (size_t)p * (DIM * DIM);
    #pragma unroll
    for (int it = 0; it < 16; ++it) {
        int idx = it * 256 + tid;
        int d = idx >> 6, e = idx & 63;
        uint32_t u = __builtin_bit_cast(uint32_t, wp[idx]);
        u = (u + 0x7FFFu + ((u >> 16) & 1u)) >> 16;   // RNE
        t[e * 66 + d] = (uint16_t)u;
    }
    __syncthreads();
    uint16_t* o = Wt + (size_t)p * (DIM * DIM);
    #pragma unroll
    for (int it = 0; it < 8; ++it) {
        int hw = (it * 256 + tid) * 2;
        int e = hw >> 6, d = hw & 63;
        uint32_t lo = t[e * 66 + d], hi = t[e * 66 + d + 1];
        *(uint32_t*)(o + hw) = lo | (hi << 16);
    }
}

// ---------- main: j-major FmFM, async double-buffered W, WMMA accumulators ----------
__global__ __launch_bounds__(256) void fmfm_wmma_kernel(
    const float* __restrict__ x, const uint16_t* __restrict__ Wt,
    float* __restrict__ out)
{
    __shared__ uint16_t ldsW[2][DIM * LSTR];   // double-buffered W tile, 2 x 9216 B

    const int tid  = threadIdx.x;
    const int wave = tid >> 5;                 // 8 waves, one 16-batch tile each
    const int lane = tid & 31;
    const int hgrp = lane >> 4;
    const int l16  = lane & 15;
    const int rowBase = (blockIdx.x * 8 + wave) * 16;
    const int s = blockIdx.y;                  // fj stripe 0..3

    // per-thread W-copy slot: 32 bytes (16 halfwords of one e-row)
    const int ce = tid >> 2, cd = (tid & 3) * 16;
    const uint32_t ldsOfs[2] = {
        (uint32_t)(uintptr_t)(&ldsW[0][ce * LSTR + cd]),
        (uint32_t)(uintptr_t)(&ldsW[1][ce * LSTR + cd]) };
    const size_t wOff = (size_t)ce * DIM + cd;

    const float* xiB = x + (size_t)(rowBase + l16) * FIELDS * DIM;
    const size_t xjRowStride = (size_t)FIELDS * DIM;

    float acc[8];
    #pragma unroll
    for (int r = 0; r < 8; ++r) acc[r] = 0.0f;

    // prologue: async-stage the first pair of this stripe into buffer 0
    int cur = 0;
    if (s + 1 < FIELDS)
        async_copy32(ldsOfs[0], Wt + (size_t)(s + 1 - 1) * (DIM * DIM) + wOff);

    for (int fj = s + 1; fj < FIELDS; fj += NSPLIT) {
        v8f C[4] = {{}, {}, {}, {}};           // U_j tile, accumulated over fi
        int p = fj - 1;                        // pair index of (0, fj)

        for (int fi = 0; fi < fj; ++fi) {
            wait_async0();                     // current pair's W resident in ldsW[cur]
            __syncthreads();                   // ...for every wave; prev buffer now free

            // issue async copy of the NEXT pair into the other buffer (overlaps compute)
            {
                int pn = -1;
                if (fi + 1 < fj)               pn = p + FIELDS - 2 - fi;       // (fi+1, fj)
                else if (fj + NSPLIT < FIELDS) pn = fj + NSPLIT - 1;           // (0, fj')
                if (pn >= 0)
                    async_copy32(ldsOfs[cur ^ 1], Wt + (size_t)pn * (DIM * DIM) + wOff);
            }

            // ---- A fragments (Xi row, f32 -> bf16 via v_perm) ----
            const float* xiRow = xiB + fi * DIM;
            BfFrag A[2];
            #pragma unroll
            for (int ks = 0; ks < 2; ++ks) {
                const int b0 = 8 * hgrp + 32 * ks;
                const int b1 = b0 + 16;
                float4 f0 = *(const float4*)(xiRow + b0);
                float4 f1 = *(const float4*)(xiRow + b0 + 4);
                float4 g0 = *(const float4*)(xiRow + b1);
                float4 g1 = *(const float4*)(xiRow + b1 + 4);
                A[ks].u[0] = pack_bf16_trunc(f0.x, f0.y);
                A[ks].u[1] = pack_bf16_trunc(f0.z, f0.w);
                A[ks].u[2] = pack_bf16_trunc(f1.x, f1.y);
                A[ks].u[3] = pack_bf16_trunc(f1.z, f1.w);
                A[ks].u[4] = pack_bf16_trunc(g0.x, g0.y);
                A[ks].u[5] = pack_bf16_trunc(g0.z, g0.w);
                A[ks].u[6] = pack_bf16_trunc(g1.x, g1.y);
                A[ks].u[7] = pack_bf16_trunc(g1.z, g1.w);
            }

            // ---- B fragments from LDS (2 x ds_load_b128 each) + WMMA accumulate ----
            const uint16_t* wbuf = ldsW[cur];
            #pragma unroll
            for (int nt = 0; nt < 4; ++nt) {
                const int n = l16 + 16 * nt;
                #pragma unroll
                for (int ks = 0; ks < 2; ++ks) {
                    BfFrag B;
                    const uint16_t* bp = wbuf + n * LSTR + 8 * hgrp + 32 * ks;
                    B.q[0] = *(const uint4*)(bp);
                    B.q[1] = *(const uint4*)(bp + 16);
                    C[nt] = __builtin_amdgcn_wmma_f32_16x16x32_bf16(
                                false, A[ks].v, false, B.v,
                                (short)0, C[nt], false, false);
                }
            }

            p += FIELDS - 2 - fi;              // pair (fi+1, fj)
            cur ^= 1;
        }

        // ---- second contraction, once per fj: <U_j[b,:], xj[b,:]> ----
        #pragma unroll
        for (int nt = 0; nt < 4; ++nt) {
            #pragma unroll
            for (int r = 0; r < 8; ++r) {
                size_t row = (size_t)(rowBase + r + 8 * hgrp);
                float xj = x[row * xjRowStride + (size_t)fj * DIM + l16 + 16 * nt];
                acc[r] = fmaf(C[nt][r], xj, acc[r]);
            }
        }
    }

    // ---- deferred butterfly reduction (masks < 16: halves never mix) ----
    #pragma unroll
    for (int r = 0; r < 8; ++r) {
        float v = acc[r];
        #pragma unroll
        for (int m = 8; m >= 1; m >>= 1)
            v += __shfl_xor(v, m);
        acc[r] = v;
    }
    if (l16 == 0) {
        #pragma unroll
        for (int r = 0; r < 8; ++r)
            atomicAdd(out + rowBase + r + 8 * hgrp, acc[r]);
    }
}

extern "C" void kernel_launch(void* const* d_in, const int* in_sizes, int n_in,
                              void* d_out, int out_size, void* d_ws, size_t ws_size,
                              hipStream_t stream) {
    const float* x = (const float*)d_in[0];   // (16384, 24, 64) f32
    const float* W = (const float*)d_in[1];   // (276, 64, 64) f32
    float* out = (float*)d_out;               // (16384, 1) f32
    uint16_t* Wt = (uint16_t*)d_ws;           // (276, 64, 64) bf16, transposed [e][d] : 2.2 MB

    hipMemsetAsync(out, 0, (size_t)out_size * sizeof(float), stream);
    w_transpose_bf16<<<NPAIRS, 256, 0, stream>>>(W, Wt);

    dim3 grid(BATCH / (16 * 8), NSPLIT);      // 128 batch-blocks x 4 fj-stripes
    fmfm_wmma_kernel<<<grid, 256, 0, stream>>>(x, Wt, out);
}
